// MHAtt_68659347193901
// MI455X (gfx1250) — matrix operand order repaired and checked by reading, not compiled, not run on hardware
//
#include <hip/hip_runtime.h>
#include <hip/hip_bf16.h>

// ---------------------------------------------------------------------------
// MHAtt with post-softmax group-prior scaling, CDNA5 (gfx1250, wave32).
// All GEMMs via v_wmma_f32_16x16x32_bf16. K/V are written by the projection
// kernels directly in WMMA B-fragment order so the flash-attention loop loads
// fragments with contiguous b128 reads. P (softmax tile) goes through a
// per-wave LDS swizzle to convert C-fragment -> A-fragment layout.
// ---------------------------------------------------------------------------

typedef __bf16 bf16;
typedef __bf16 bf16x16 __attribute__((ext_vector_type(16)));
typedef float  f32x8   __attribute__((ext_vector_type(8)));

#define BATCH   4
#define S_LEN   1024
#define HID     1024
#define NHEAD   16
#define DHEAD   64
#define MTOT    (BATCH * S_LEN)        // 4096 rows for every GEMM

union FragU {           // one wave-lane slice of a 16x32 (A) or 32x16 (B) tile
    uint4   q[2];       // 2 x 16B contiguous chunks
    bf16x16 v;
};

// epilogue modes for the projection GEMM
#define MODE_QHEAD  0   // bf16, head-major [B,NH,S,DH]  (A-frag friendly)
#define MODE_KFRAG  1   // bf16, B-fragment-major, contraction dim = DH (64)
#define MODE_VFRAG  2   // bf16, B-fragment-major, contraction dim = S (1024)
#define MODE_OUTF32 3   // f32 row-major [M,N] (final output)

// convert 8 consecutive f32 to one 16B packet of bf16
__device__ __forceinline__ uint4 cvt8_bf16(float4 a, float4 b) {
    union { bf16 h[8]; uint4 u; } o;
    o.h[0] = (bf16)a.x; o.h[1] = (bf16)a.y; o.h[2] = (bf16)a.z; o.h[3] = (bf16)a.w;
    o.h[4] = (bf16)b.x; o.h[5] = (bf16)b.y; o.h[6] = (bf16)b.z; o.h[7] = (bf16)b.w;
    return o.u;
}

// ---------------------------------------------------------------------------
// Y[m,n] = sum_k X[m,k] * W[n,k] + bias[n], M=4096, N=K=1024.
// Block = 256 threads (8 waves), tile 128x64, K-step 32.
// LDS tiles are stored pre-swizzled in fragment order; staging moves whole
// 8-element chunks (2x global b128 + 1x ds_store_b128) so loads batch up.
// ---------------------------------------------------------------------------
template <int MODE>
__global__ __launch_bounds__(256)
void gemm_proj_kernel(const float* __restrict__ X, const float* __restrict__ W,
                      const float* __restrict__ bias, void* __restrict__ out)
{
    __shared__ bf16 sA[128 * 32];  // fragment-order A tile (8KB)
    __shared__ bf16 sB[32 * 64];   // fragment-order B tile (4KB)

    const int tid  = threadIdx.x;
    const int lane = tid & 31;
    const int wave = tid >> 5;
    const int m0   = blockIdx.y * 128;
    const int n0   = blockIdx.x * 64;

    uint4* sAq = (uint4*)sA;
    uint4* sBq = (uint4*)sB;

    f32x8 acc[4] = {};   // wave owns rows [m0+16*wave, +16), cols [n0, n0+64)

    for (int kt = 0; kt < HID; kt += 32) {
        __syncthreads();
        // prefetch next K-tile (speculative; dropped if OOB)
        __builtin_prefetch(&X[(m0 + (tid >> 1)) * HID + kt + 32], 0, 1);
        __builtin_prefetch(&W[(n0 + (tid >> 2)) * HID + kt + 32], 0, 1);

        // ---- stage A: 128 rows x 32 k = 512 chunks of 8, 2 per thread ----
        #pragma unroll
        for (int e = 0; e < 2; ++e) {
            int cid = e * 256 + tid;
            int r = cid >> 2, g = cid & 3;          // k = 8g .. 8g+7
            const float4* xs = (const float4*)&X[(m0 + r) * HID + kt + 8 * g];
            float4 x0 = xs[0], x1 = xs[1];
            // chunk (hi = g&1, c = g>>1) of lane (r&15)+16*hi in wave r>>4
            sAq[((r >> 4) * 32 + (r & 15) + 16 * (g & 1)) * 2 + (g >> 1)] =
                cvt8_bf16(x0, x1);
        }
        // ---- stage B: 32 k x 64 cols = 256 chunks of 8, 1 per thread ----
        {
            int n = tid >> 2, g = tid & 3;
            const float4* wsrc = (const float4*)&W[(n0 + n) * HID + kt + 8 * g];
            float4 w0 = wsrc[0], w1 = wsrc[1];
            // chunk (hi = g&1, c = g>>1) of lane (n&15)+16*hi in n-block n>>4
            sBq[((n >> 4) * 32 + (n & 15) + 16 * (g & 1)) * 2 + (g >> 1)] =
                cvt8_bf16(w0, w1);
        }
        __syncthreads();

        FragU fa;
        fa.q[0] = sAq[(wave * 32 + lane) * 2 + 0];
        fa.q[1] = sAq[(wave * 32 + lane) * 2 + 1];
        #pragma unroll
        for (int t = 0; t < 4; ++t) {
            FragU fb;
            fb.q[0] = sBq[(t * 32 + lane) * 2 + 0];
            fb.q[1] = sBq[(t * 32 + lane) * 2 + 1];
            acc[t] = __builtin_amdgcn_wmma_f32_16x16x32_bf16(
                         false, fa.v, false, fb.v, (short)0, acc[t], false, false);
        }
    }

    // ---- epilogue: bias + scatter into the requested layout ----
    const int hi  = lane >> 4;      // C layout: row = i + 8*hi, col = 16t + lane%16
    const int c16 = lane & 15;
    bf16*  ob = (bf16*)out;
    float* of = (float*)out;

    #pragma unroll
    for (int t = 0; t < 4; ++t) {
        int n  = n0 + 16 * t + c16;
        float bs = bias[n];
        #pragma unroll
        for (int i = 0; i < 8; ++i) {
            int m   = m0 + 16 * wave + i + 8 * hi;
            float y = acc[t][i] + bs;
            int b = m >> 10, s = m & 1023;       // m = b*S + s
            int h = n >> 6,  d = n & 63;         // n = h*DH + d
            if constexpr (MODE == MODE_QHEAD) {
                ob[((b * NHEAD + h) * S_LEN + s) * DHEAD + d] = (bf16)y;
            } else if constexpr (MODE == MODE_KFRAG) {
                // B-frag over (K=dh 64, N=key 1024): per-(b,h) 64*1024 elems
                int base = (b * NHEAD + h) * (DHEAD * S_LEN);
                int kb = d >> 5, nb = s >> 4, kin = d & 31;
                int ln = (s & 15) + ((kin & 8) ? 16 : 0);
                int c  = kin >> 4;
                ob[base + (((kb * 64 + nb) * 32 + ln) * 2 + c) * 8 + (kin & 7)] = (bf16)y;
            } else if constexpr (MODE == MODE_VFRAG) {
                // B-frag over (K=key 1024, N=dh 64)
                int base = (b * NHEAD + h) * (S_LEN * DHEAD);
                int kb = s >> 5, nb = d >> 4, kin = s & 31;
                int ln = (d & 15) + ((kin & 8) ? 16 : 0);
                int c  = kin >> 4;
                ob[base + (((kb * 4 + nb) * 32 + ln) * 2 + c) * 8 + (kin & 7)] = (bf16)y;
            } else { // MODE_OUTF32
                of[m * HID + n] = y;
            }
        }
    }
}

// ---------------------------------------------------------------------------
// Fused flash attention: per block (b, h, 64-row q tile), 4 waves x 16 q rows.
// Online softmax; group_prob scales the numerator only (denominator l = sum exp,
// matching softmax(scores) * gp). mask==true -> score = -1e9 (per key column).
// ---------------------------------------------------------------------------
__global__ __launch_bounds__(128)
void flash_attn_kernel(const bf16* __restrict__ qh, const bf16* __restrict__ kf,
                       const bf16* __restrict__ vf, const unsigned char* __restrict__ mask,
                       const float* __restrict__ gp, float* __restrict__ attOut)
{
    __shared__ uint4 pbuf[4 * 64];  // per-wave 16x32 bf16 P tile, A-frag order

    const int lane  = threadIdx.x & 31;
    const int wave  = threadIdx.x >> 5;
    const int bh    = blockIdx.x >> 4;          // b*NH + h
    const int qtile = blockIdx.x & 15;
    const int b     = bh >> 4;
    const int h     = bh & 15;
    const int q0    = qtile * 64 + wave * 16;   // this wave's first q row

    const int hi  = lane >> 4;
    const int c16 = lane & 15;

    // Q A-fragments (16 rows x DH=64 -> two K=32 fragments), direct from global
    const uint4* qp = (const uint4*)(qh + (bh * S_LEN + q0 + c16) * DHEAD);
    FragU aq0, aq1;
    {
        int k0 = hi ? 8 : 0;                    // A-frag chunk K offsets
        aq0.q[0] = qp[(k0) >> 3];
        aq0.q[1] = qp[(k0 + 16) >> 3];
        aq1.q[0] = qp[(32 + k0) >> 3];
        aq1.q[1] = qp[(32 + k0 + 16) >> 3];
    }

    const uint4* kfb = (const uint4*)(kf + bh * (DHEAD * S_LEN));
    const uint4* vfb = (const uint4*)(vf + bh * (S_LEN * DHEAD));

    float mrow[8], lrow[8];
    f32x8 acc[4] = {};
    #pragma unroll
    for (int i = 0; i < 8; ++i) { mrow[i] = -3.0e38f; lrow[i] = 0.0f; }
    const float scale = 0.125f;                 // 1/sqrt(DH)

    for (int kt = 0; kt < S_LEN; kt += 32) {
        const int kt16 = kt >> 4;

        // ---- batch-load group_prob for this 16x32 tile (16 loads in flight) --
        float g0a[8], g1a[8];
        #pragma unroll
        for (int i = 0; i < 8; ++i) {
            const float* gprow = &gp[(b * S_LEN + q0 + (i + 8 * hi)) * S_LEN + kt];
            g0a[i] = gprow[c16];
            g1a[i] = gprow[16 + c16];
        }

        f32x8 sc[2];
        // ---- scores: 16 q x 32 keys, contraction over DH (2 WMMA chains) ----
        #pragma unroll
        for (int tt = 0; tt < 2; ++tt) {
            FragU fb0, fb1;
            int nb = kt16 + tt;
            fb0.q[0] = kfb[((0  + nb) * 32 + lane) * 2 + 0];
            fb0.q[1] = kfb[((0  + nb) * 32 + lane) * 2 + 1];
            fb1.q[0] = kfb[((64 + nb) * 32 + lane) * 2 + 0];
            fb1.q[1] = kfb[((64 + nb) * 32 + lane) * 2 + 1];
            f32x8 c = {};
            c = __builtin_amdgcn_wmma_f32_16x16x32_bf16(false, aq0.v, false, fb0.v,
                                                        (short)0, c, false, false);
            c = __builtin_amdgcn_wmma_f32_16x16x32_bf16(false, aq1.v, false, fb1.v,
                                                        (short)0, c, false, false);
            int key = kt + 16 * tt + c16;       // whole fragment column = one key
            bool mk = mask[b * S_LEN + key] != 0;
            #pragma unroll
            for (int i = 0; i < 8; ++i)
                sc[tt][i] = mk ? -1.0e9f : c[i] * scale;
        }

        // ---- online softmax update + gp-scaled P into LDS (A-frag order) ----
        bf16* pb = (bf16*)(pbuf + wave * 64);
        #pragma unroll
        for (int i = 0; i < 8; ++i) {
            float vmax = fmaxf(sc[0][i], sc[1][i]);
            vmax = fmaxf(vmax, __shfl_xor(vmax, 1));
            vmax = fmaxf(vmax, __shfl_xor(vmax, 2));
            vmax = fmaxf(vmax, __shfl_xor(vmax, 4));
            vmax = fmaxf(vmax, __shfl_xor(vmax, 8));
            float nm = fmaxf(mrow[i], vmax);

            float p0 = __expf(sc[0][i] - nm);
            float p1 = __expf(sc[1][i] - nm);
            float ls = p0 + p1;                  // denominator has NO gp factor
            ls += __shfl_xor(ls, 1);
            ls += __shfl_xor(ls, 2);
            ls += __shfl_xor(ls, 4);
            ls += __shfl_xor(ls, 8);

            float f = __expf(mrow[i] - nm);
            lrow[i] = lrow[i] * f + ls;
            mrow[i] = nm;
            #pragma unroll
            for (int t = 0; t < 4; ++t) acc[t][i] *= f;

            int r = i + 8 * hi;                  // C-layout row of this element
            // scatter into A-frag order: lane_a = r + 16*((kin>>3)&1), c=kin>>4
            int ha = (c16 >> 3) & 1;
            pb[((r + 16 * ha) * 2 + 0) * 8 + (c16 & 7)] = (bf16)(p0 * g0a[i]);
            pb[((r + 16 * ha) * 2 + 1) * 8 + (c16 & 7)] = (bf16)(p1 * g1a[i]);
        }

        // ---- P(16x32) @ V(32x64): 4 WMMAs, V frags straight from global ----
        FragU pa;
        const uint4* pq = pbuf + wave * 64;
        pa.q[0] = pq[lane * 2 + 0];
        pa.q[1] = pq[lane * 2 + 1];
        int kb = kt >> 5;
        #pragma unroll
        for (int t = 0; t < 4; ++t) {
            FragU fv;
            fv.q[0] = vfb[((kb * 4 + t) * 32 + lane) * 2 + 0];
            fv.q[1] = vfb[((kb * 4 + t) * 32 + lane) * 2 + 1];
            acc[t] = __builtin_amdgcn_wmma_f32_16x16x32_bf16(
                         false, pa.v, false, fv.v, (short)0, acc[t], false, false);
        }
    }

    // ---- normalize and write concat-head layout [B,S,H] (f32) ----
    #pragma unroll
    for (int i = 0; i < 8; ++i) {
        int r = i + 8 * hi;
        float inv = 1.0f / lrow[i];
        #pragma unroll
        for (int t = 0; t < 4; ++t)
            attOut[(b * S_LEN + q0 + r) * HID + h * DHEAD + 16 * t + c16] = acc[t][i] * inv;
    }
}

// ---------------------------------------------------------------------------
extern "C" void kernel_launch(void* const* d_in, const int* in_sizes, int n_in,
                              void* d_out, int out_size, void* d_ws, size_t ws_size,
                              hipStream_t stream)
{
    (void)in_sizes; (void)n_in; (void)out_size; (void)ws_size;

    const float* v  = (const float*)d_in[0];
    const float* k  = (const float*)d_in[1];
    const float* q  = (const float*)d_in[2];
    const unsigned char* mask = (const unsigned char*)d_in[3]; // jax bool, 1B/elem
    const float* gp = (const float*)d_in[4];
    const float* Wq = (const float*)d_in[5];  const float* bq = (const float*)d_in[6];
    const float* Wk = (const float*)d_in[7];  const float* bk = (const float*)d_in[8];
    const float* Wv = (const float*)d_in[9];  const float* bv = (const float*)d_in[10];
    const float* Wm = (const float*)d_in[11]; const float* bm = (const float*)d_in[12];

    // workspace: qh (bf16 8MB) | kfrag (bf16 8MB) | vfrag (bf16 8MB) | attOut (f32 16MB)
    char* ws = (char*)d_ws;
    bf16*  qh = (bf16*)(ws);
    bf16*  kf = (bf16*)(ws + (size_t)8  * 1024 * 1024);
    bf16*  vf = (bf16*)(ws + (size_t)16 * 1024 * 1024);
    float* ao = (float*)(ws + (size_t)24 * 1024 * 1024);

    dim3 ggrid(HID / 64, MTOT / 128);  // (16, 32)
    dim3 gblk(256);

    gemm_proj_kernel<MODE_QHEAD ><<<ggrid, gblk, 0, stream>>>(q, Wq, bq, (void*)qh);
    gemm_proj_kernel<MODE_KFRAG ><<<ggrid, gblk, 0, stream>>>(k, Wk, bk, (void*)kf);
    gemm_proj_kernel<MODE_VFRAG ><<<ggrid, gblk, 0, stream>>>(v, Wv, bv, (void*)vf);

    flash_attn_kernel<<<dim3(BATCH * NHEAD * (S_LEN / 64)), dim3(128), 0, stream>>>(
        qh, kf, vf, mask, gp, ao);

    gemm_proj_kernel<MODE_OUTF32><<<ggrid, gblk, 0, stream>>>(ao, Wm, bm, d_out);
}
